// TransformerBlock_15496242004504
// MI455X (gfx1250) — matrix operand order repaired and checked by reading, not compiled, not run on hardware
//
#include <hip/hip_runtime.h>
#include <hip/hip_bf16.h>

// ---------------------------------------------------------------------------
// Transformer block for MI455X (gfx1250, wave32, WMMA, async-to-LDS).
//   x = x + attn(rmsnorm(x)) ; x = x + swiglu(rmsnorm(x))
// GEMMs + flash attention run on v_wmma_f32_16x16x32_f16 (f32 accumulate).
// Weight matrices are converted+transposed once per launch so every LDS stage
// is a contiguous 16B chunk -> global_load_async_to_lds_b128, double-buffered.
// ---------------------------------------------------------------------------

#define DIM      2048
#define FFN      8192
#define SEQ      2048
#define BATCH    2
#define NTOK     (BATCH * SEQ)      // 4096
#define NHEADS   16
#define HDIM     128                // DIM / NHEADS
#define EPS      1e-6f

typedef __attribute__((ext_vector_type(16))) _Float16 v16h;
typedef __attribute__((ext_vector_type(8)))  float    v8f;
typedef __attribute__((ext_vector_type(4)))  int      v4i;

#define HAVE_ASYNC_LDS __has_builtin(__builtin_amdgcn_global_load_async_to_lds_b128)

__device__ __forceinline__ void stage16(void* lds_dst, const void* gsrc) {
#if HAVE_ASYNC_LDS
    __builtin_amdgcn_global_load_async_to_lds_b128(
        (__attribute__((address_space(1))) v4i*)gsrc,
        (__attribute__((address_space(3))) v4i*)lds_dst, 0, 0);
#else
    *(uint4*)lds_dst = *(const uint4*)gsrc;
#endif
}

__device__ __forceinline__ void stage_wait() {
#if HAVE_ASYNC_LDS
#if __has_builtin(__builtin_amdgcn_s_wait_asynccnt)
    __builtin_amdgcn_s_wait_asynccnt(0);
#else
    asm volatile("s_wait_asynccnt 0x0" ::: "memory");
#endif
#endif
}

// --------------------- f32 -> f16 transposed convert -----------------------
// in: w[K][N] f32, out: o[N][K] f16.  32x32 tiles through LDS.
__global__ __launch_bounds__(256) void cvt_f16t_kernel(const float* __restrict__ w,
                                                       _Float16* __restrict__ o,
                                                       int Kd, int Nd) {
    __shared__ _Float16 tile[32][33];
    const int bk = blockIdx.y * 32, bn = blockIdx.x * 32;
    const int tx = threadIdx.x & 31, ty = threadIdx.x >> 5;  // 32 x 8
#pragma unroll
    for (int r = 0; r < 32; r += 8)
        tile[ty + r][tx] = (_Float16)w[(size_t)(bk + ty + r) * Nd + (bn + tx)];
    __syncthreads();
#pragma unroll
    for (int r = 0; r < 32; r += 8)
        o[(size_t)(bn + ty + r) * Kd + (bk + tx)] = tile[tx][ty + r];
}

// ------------------------------- RMSNorm -----------------------------------
__global__ __launch_bounds__(256) void rmsnorm_f16_kernel(const float* __restrict__ x,
                                                          const float* __restrict__ g,
                                                          _Float16* __restrict__ y) {
    __shared__ float red[8];
    __shared__ float rms_s;
    const int row = blockIdx.x;
    const float* xr = x + (size_t)row * DIM;
    float ss = 0.f;
    for (int i = threadIdx.x; i < DIM; i += 256) {
        float v = xr[i];
        ss += v * v;
    }
    for (int off = 16; off; off >>= 1) ss += __shfl_xor(ss, off, 32);
    if ((threadIdx.x & 31) == 0) red[threadIdx.x >> 5] = ss;
    __syncthreads();
    if (threadIdx.x == 0) {
        float t = 0.f;
        for (int i = 0; i < 8; ++i) t += red[i];
        rms_s = rsqrtf(t / (float)DIM + EPS);
    }
    __syncthreads();
    const float rms = rms_s;
    _Float16* yr = y + (size_t)row * DIM;
    for (int i = threadIdx.x; i < DIM; i += 256)
        yr[i] = (_Float16)(xr[i] * rms * g[i]);
}

// ------------------------------- WMMA GEMM ---------------------------------
// C[M,N] = A16[M,K] * Bt16[N,K]^T.  Block = 8 waves, tile 64x128, K-step 32.
// Double-buffered LDS; tiles stream via async b128 copies overlapped with the
// WMMA work on the other buffer (split-wait on ASYNCcnt + single barrier).
enum { EP_F16 = 0, EP_F32_RESID = 1, EP_SILU_MUL_F16 = 2, EP_F16_VT = 3 };

template <int EPI>
__global__ __launch_bounds__(256) void gemm_wmma_kernel(
    const _Float16* __restrict__ A, const _Float16* __restrict__ Bt,
    void* __restrict__ Out, const void* __restrict__ Aux,
    int M, int N, int K) {
    __shared__ _Float16 sA[2][64 * 40];    // [m][k], row stride 80B
    __shared__ _Float16 sB[2][128 * 40];   // [n][k], row stride 80B

    const int tid = threadIdx.x;
    const int wid = tid >> 5, lane = tid & 31;
    const int wm  = wid & 3;            // 4 waves along M
    const int wn  = wid >> 2;           // 2 waves along N
    const int m0  = blockIdx.y * 64;
    const int n0  = blockIdx.x * 128;
    const int h   = lane >> 4;
    const int ln  = lane & 15;

    v8f acc[4];
#pragma unroll
    for (int j = 0; j < 4; ++j)
#pragma unroll
        for (int r = 0; r < 8; ++r) acc[j][r] = 0.f;

    const int arow = tid >> 2, achk = tid & 3;   // A: 64 rows x 4 chunks
    auto stage_tiles = [&](int buf, int k0) {
        // A tile 64x32: one 16B chunk per thread.
        stage16(&sA[buf][arow * 40 + achk * 8],
                A + (size_t)(m0 + arow) * K + k0 + achk * 8);
        // B tile 128x32 (pre-transposed weights): two 16B chunks per thread.
#pragma unroll
        for (int qq = 0; qq < 2; ++qq) {
            const int i = tid + qq * 256;
            const int brow = i >> 2, bchk = i & 3;
            stage16(&sB[buf][brow * 40 + bchk * 8],
                    Bt + (size_t)(n0 + brow) * K + k0 + bchk * 8);
        }
    };

    const int nk = K >> 5;
    stage_tiles(0, 0);
    for (int ki = 0; ki < nk; ++ki) {
        stage_wait();          // previous stage landed in LDS
        __syncthreads();       // visible to all waves; prior reads retired
        if (ki + 1 < nk) stage_tiles((ki + 1) & 1, (ki + 1) << 5);

        const _Float16* cA = &sA[ki & 1][0];
        const _Float16* cB = &sB[ki & 1][0];

        // A fragment (k = 16h + s, same order as B): two contiguous 16B ds loads.
        union { unsigned u[8]; v16h v; } fa;
        const unsigned* pa = (const unsigned*)(cA + (wm * 16 + ln) * 40);
#pragma unroll
        for (int r = 0; r < 8; ++r) {
            const int kp = h * 16 + 2 * r;
            fa.u[r] = pa[kp >> 1];
        }
#pragma unroll
        for (int j = 0; j < 4; ++j) {
            union { unsigned u[8]; v16h v; } fb;
            const unsigned* pb = (const unsigned*)(cB + (wn * 64 + j * 16 + ln) * 40);
#pragma unroll
            for (int r = 0; r < 8; ++r) fb.u[r] = pb[h * 8 + r];
            acc[j] = __builtin_amdgcn_wmma_f32_16x16x32_f16(
                false, fa.v, false, fb.v, (short)0, acc[j], false, false);
        }
    }

    // Epilogue: C layout — VGPR r holds M = r + 8h, lane ln = N column.
#pragma unroll
    for (int j = 0; j < 4; ++j) {
        const int n = n0 + wn * 64 + j * 16 + ln;
#pragma unroll
        for (int r = 0; r < 8; ++r) {
            const int m = m0 + wm * 16 + h * 8 + r;
            const size_t idx = (size_t)m * N + n;
            const float val = acc[j][r];
            if (EPI == EP_F16) {
                ((_Float16*)Out)[idx] = (_Float16)val;
            } else if (EPI == EP_F32_RESID) {
                ((float*)Out)[idx] = val + ((const float*)Aux)[idx];
            } else if (EPI == EP_SILU_MUL_F16) {
                const float gt = (float)((const _Float16*)Aux)[idx];
                const float s  = gt / (1.f + __expf(-gt));
                ((_Float16*)Out)[idx] = (_Float16)(s * val);
            } else {  // EP_F16_VT: write V transposed per head: [b][head][d][t]
                const int t    = m & (SEQ - 1);
                const int bb   = m >> 11;           // m / SEQ
                const int head = n >> 7;            // n / HDIM
                const int d    = n & (HDIM - 1);
                ((_Float16*)Out)[(((size_t)(bb * NHEADS + head) * HDIM) + d) * SEQ + t] =
                    (_Float16)val;
            }
        }
    }
}

// ------------------------- WMMA flash attention ----------------------------
// One wave per (b, head, 16-query tile).  Q lives in registers as 4 A-frags.
// K fragments load straight from global ([token][d] == the [n][k] B layout
// for Q.K^T); V fragments load from the pre-transposed Vt buffer.  Online
// softmax reductions are 16-lane shfl_xor butterflies (rows align between the
// S accumulator and output accumulator layouts).  16 WMMAs per 32-key tile.
__global__ __launch_bounds__(256) void flash_wmma_kernel(
    const _Float16* __restrict__ q,   // [NTOK][DIM]
    const _Float16* __restrict__ k,   // [NTOK][DIM]
    const _Float16* __restrict__ vt,  // [BATCH][NHEADS][HDIM][SEQ]
    _Float16* __restrict__ out) {     // [NTOK][DIM]
    __shared__ _Float16 sP[8][16 * 40];   // per-wave P redistribute tile

    const int tid = threadIdx.x, wid = tid >> 5, lane = tid & 31;
    const int h = lane >> 4, ln = lane & 15;
    const int task = blockIdx.x * 8 + wid;       // 4096 tasks
    const int qt   = task & 127;
    const int head = (task >> 7) & (NHEADS - 1);
    const int b    = task >> 11;
    const int qt0  = qt * 16;
    const size_t hbase = (size_t)(b * SEQ) * DIM + head * HDIM;
    const size_t vbase = ((size_t)(b * NHEADS + head)) * HDIM * SEQ;
    _Float16* sp = &sP[wid][0];

    union F { unsigned u[8]; v16h v; };
    // Q fragments: 4 K-steps over head_dim.
    F qf[4];
#pragma unroll
    for (int s = 0; s < 4; ++s) {
        const unsigned* p32 =
            (const unsigned*)(q + hbase + (size_t)(qt0 + ln) * DIM + s * 32);
#pragma unroll
        for (int r = 0; r < 8; ++r) {
            const int kp = h * 16 + 2 * r;
            qf[s].u[r] = p32[kp >> 1];
        }
    }

    float mrow[8], lrow[8];
    v8f acc[8];
#pragma unroll
    for (int r = 0; r < 8; ++r) { mrow[r] = -1e30f; lrow[r] = 0.f; }
#pragma unroll
    for (int d = 0; d < 8; ++d)
#pragma unroll
        for (int r = 0; r < 8; ++r) acc[d][r] = 0.f;

    const float scale = 0.08838834764831845f;  // 1/sqrt(128)

    for (int k0 = 0; k0 < qt0 + 16; k0 += 32) {
        // ---- S = Q . K^T for 32 keys (2 sub-tiles x 4 d-steps) ----
        v8f S[2];
#pragma unroll
        for (int j = 0; j < 2; ++j) {
#pragma unroll
            for (int r = 0; r < 8; ++r) S[j][r] = 0.f;
#pragma unroll
            for (int s = 0; s < 4; ++s) {
                F kf;
                const unsigned* p32 = (const unsigned*)(
                    k + hbase + (size_t)(k0 + j * 16 + ln) * DIM + s * 32);
#pragma unroll
                for (int r = 0; r < 8; ++r) kf.u[r] = p32[h * 8 + r];
                S[j] = __builtin_amdgcn_wmma_f32_16x16x32_f16(
                    false, qf[s].v, false, kf.v, (short)0, S[j], false, false);
            }
        }
        // ---- scale + causal mask ----
#pragma unroll
        for (int r = 0; r < 8; ++r) {
            const int qrow = qt0 + h * 8 + r;
#pragma unroll
            for (int j = 0; j < 2; ++j) {
                const int key = k0 + j * 16 + ln;
                float sv = S[j][r] * scale;
                if (key > qrow) sv = -1e30f;
                S[j][r] = sv;
            }
        }
        // ---- online softmax (per-row over 16 lanes) ----
        float corr[8];
#pragma unroll
        for (int r = 0; r < 8; ++r) {
            float t = fmaxf(S[0][r], S[1][r]);
            for (int o2 = 8; o2; o2 >>= 1) t = fmaxf(t, __shfl_xor(t, o2, 32));
            const float mn = fmaxf(mrow[r], t);
            corr[r] = __expf(mrow[r] - mn);
            mrow[r] = mn;
            const float p0 = __expf(S[0][r] - mn);
            const float p1 = __expf(S[1][r] - mn);
            S[0][r] = p0;
            S[1][r] = p1;
            float rs = p0 + p1;
            for (int o2 = 8; o2; o2 >>= 1) rs += __shfl_xor(rs, o2, 32);
            lrow[r] = lrow[r] * corr[r] + rs;
        }
        // ---- P -> LDS (C layout -> A layout redistribute) ----
#pragma unroll
        for (int j = 0; j < 2; ++j)
#pragma unroll
            for (int r = 0; r < 8; ++r)
                sp[(h * 8 + r) * 40 + j * 16 + ln] = (_Float16)S[j][r];
        // ---- rescale output accumulators ----
#pragma unroll
        for (int d = 0; d < 8; ++d)
#pragma unroll
            for (int r = 0; r < 8; ++r) acc[d][r] *= corr[r];
        // ---- P fragment ----
        F pf;
        const unsigned* pp = (const unsigned*)(sp + ln * 40);
#pragma unroll
        for (int r = 0; r < 8; ++r) {
            const int kp = h * 16 + 2 * r;
            pf.u[r] = pp[kp >> 1];
        }
        // ---- acc += P . V (8 d-subtiles, K = 32 keys) ----
#pragma unroll
        for (int d = 0; d < 8; ++d) {
            F vf;
            const unsigned* p32 = (const unsigned*)(
                vt + vbase + (size_t)(d * 16 + ln) * SEQ + k0);
#pragma unroll
            for (int r = 0; r < 8; ++r) vf.u[r] = p32[h * 8 + r];
            acc[d] = __builtin_amdgcn_wmma_f32_16x16x32_f16(
                false, pf.v, false, vf.v, (short)0, acc[d], false, false);
        }
    }
    // ---- normalize + store ----
#pragma unroll
    for (int d = 0; d < 8; ++d)
#pragma unroll
        for (int r = 0; r < 8; ++r) {
            const int qrow = qt0 + h * 8 + r;
            out[hbase + (size_t)qrow * DIM + d * 16 + ln] =
                (_Float16)(acc[d][r] / lrow[r]);
        }
}

// ------------------------------- driver ------------------------------------
extern "C" void kernel_launch(void* const* d_in, const int* in_sizes, int n_in,
                              void* d_out, int out_size, void* d_ws, size_t ws_size,
                              hipStream_t stream) {
    (void)in_sizes; (void)n_in; (void)out_size; (void)ws_size;

    const float* x      = (const float*)d_in[0];
    const float* wq     = (const float*)d_in[1];
    const float* wk     = (const float*)d_in[2];
    const float* wv     = (const float*)d_in[3];
    const float* wo     = (const float*)d_in[4];
    const float* w_gate = (const float*)d_in[5];
    const float* w_up   = (const float*)d_in[6];
    const float* w_down = (const float*)d_in[7];
    const float* g_attn = (const float*)d_in[8];
    const float* g_ffn  = (const float*)d_in[9];
    float* out = (float*)d_out;

    size_t off = 0;
    auto alloc = [&](size_t bytes) -> char* {
        char* p = (char*)d_ws + off;
        off += (bytes + 255) & ~(size_t)255;
        return p;
    };
    // Transposed f16 weights: [N][K].
    _Float16* wqT = (_Float16*)alloc((size_t)DIM * DIM * 2);
    _Float16* wkT = (_Float16*)alloc((size_t)DIM * DIM * 2);
    _Float16* wvT = (_Float16*)alloc((size_t)DIM * DIM * 2);
    _Float16* woT = (_Float16*)alloc((size_t)DIM * DIM * 2);
    _Float16* wgT = (_Float16*)alloc((size_t)DIM * FFN * 2);
    _Float16* wuT = (_Float16*)alloc((size_t)DIM * FFN * 2);
    _Float16* wdT = (_Float16*)alloc((size_t)FFN * DIM * 2);
    _Float16* xn16 = (_Float16*)alloc((size_t)NTOK * DIM * 2);
    _Float16* q16  = (_Float16*)alloc((size_t)NTOK * DIM * 2);
    _Float16* k16  = (_Float16*)alloc((size_t)NTOK * DIM * 2);
    _Float16* vt16 = (_Float16*)alloc((size_t)NTOK * DIM * 2);
    _Float16* ao16 = (_Float16*)alloc((size_t)NTOK * DIM * 2);
    _Float16* gb16 = (_Float16*)alloc((size_t)NTOK * FFN * 2);

    // 1) convert + transpose weights (w[K][N] -> wT[N][K], f16)
    dim3 tQKV(DIM / 32, DIM / 32);
    cvt_f16t_kernel<<<tQKV, 256, 0, stream>>>(wq, wqT, DIM, DIM);
    cvt_f16t_kernel<<<tQKV, 256, 0, stream>>>(wk, wkT, DIM, DIM);
    cvt_f16t_kernel<<<tQKV, 256, 0, stream>>>(wv, wvT, DIM, DIM);
    cvt_f16t_kernel<<<tQKV, 256, 0, stream>>>(wo, woT, DIM, DIM);
    dim3 tUP(FFN / 32, DIM / 32);
    cvt_f16t_kernel<<<tUP, 256, 0, stream>>>(w_gate, wgT, DIM, FFN);
    cvt_f16t_kernel<<<tUP, 256, 0, stream>>>(w_up, wuT, DIM, FFN);
    dim3 tDN(DIM / 32, FFN / 32);
    cvt_f16t_kernel<<<tDN, 256, 0, stream>>>(w_down, wdT, FFN, DIM);

    // 2) attn pre-norm
    rmsnorm_f16_kernel<<<NTOK, 256, 0, stream>>>(x, g_attn, xn16);

    // 3) Q/K/V projections; V writes per-head transposed layout for flash PV.
    dim3 gQKV(DIM / 128, NTOK / 64);
    gemm_wmma_kernel<EP_F16><<<gQKV, 256, 0, stream>>>(xn16, wqT, q16, nullptr,
                                                       NTOK, DIM, DIM);
    gemm_wmma_kernel<EP_F16><<<gQKV, 256, 0, stream>>>(xn16, wkT, k16, nullptr,
                                                       NTOK, DIM, DIM);
    gemm_wmma_kernel<EP_F16_VT><<<gQKV, 256, 0, stream>>>(xn16, wvT, vt16, nullptr,
                                                          NTOK, DIM, DIM);

    // 4) WMMA flash attention: 4096 wave-tasks, 8 waves/block.
    flash_wmma_kernel<<<512, 256, 0, stream>>>(q16, k16, vt16, ao16);

    // 5) output projection + residual: out = x + ao @ wo
    gemm_wmma_kernel<EP_F32_RESID><<<gQKV, 256, 0, stream>>>(ao16, woT, out, x,
                                                             NTOK, DIM, DIM);

    // 6) ffn pre-norm
    rmsnorm_f16_kernel<<<NTOK, 256, 0, stream>>>(out, g_ffn, xn16);

    // 7) gate, then gb = silu(gate) * up, fused in the up-GEMM epilogue.
    dim3 gFFN(FFN / 128, NTOK / 64);
    gemm_wmma_kernel<EP_F16><<<gFFN, 256, 0, stream>>>(xn16, wgT, gb16, nullptr,
                                                       NTOK, FFN, DIM);
    gemm_wmma_kernel<EP_SILU_MUL_F16><<<gFFN, 256, 0, stream>>>(
        xn16, wuT, gb16, gb16, NTOK, FFN, DIM);

    // 8) down projection + residual: out += gb @ w_down (K = 8192)
    gemm_wmma_kernel<EP_F32_RESID><<<gQKV, 256, 0, stream>>>(gb16, wdT, out, out,
                                                             NTOK, DIM, FFN);
}
